// NCAAttention_17437567222040
// MI455X (gfx1250) — matrix operand order repaired and checked by reading, not compiled, not run on hardware
//
#include <hip/hip_runtime.h>
#include <hip/hip_bf16.h>
#include <math.h>

typedef __attribute__((ext_vector_type(8)))  float   v8f;
typedef __attribute__((ext_vector_type(16))) __bf16  v16bf;
typedef __attribute__((ext_vector_type(8)))  __bf16  v8bf;

#define B_     8
#define H_     128
#define W_     128
#define CH_    64
#define OC_    64
#define NCAC_  128
#define HID_   64
#define KIN_   640        // 5 * NCAC_
#define WAVES_ 4
#define FPAD_  136        // feat row stride in halves (272B, 16B aligned, bank-skewed)
#define HPAD_  72         // hid  row stride in halves (144B)

// ---------------------------------------------------------------------------
// Kernel 1: context path.  attended == v exactly (softmax weights sum to 1,
// v has no spatial dependence), so c2 = c + LN(v @ Wo + bo + c).
// ---------------------------------------------------------------------------
__global__ __launch_bounds__(512) void ctx_kernel(
    const float* __restrict__ c,
    const float* __restrict__ Wv, const float* __restrict__ bv,
    const float* __restrict__ Wo, const float* __restrict__ bo,
    const float* __restrict__ ln_g, const float* __restrict__ ln_b,
    float* __restrict__ c2ws, float* __restrict__ c2out)
{
    __shared__ float vsh[B_][OC_];
    __shared__ float tsh[B_][OC_];
    const int t = threadIdx.x;
    const int b = t >> 6;
    const int o = t & 63;

    float v = bv[o];
    for (int i = 0; i < OC_; ++i) v = fmaf(c[b * OC_ + i], Wv[i * OC_ + o], v);
    vsh[b][o] = v;
    __syncthreads();

    float tv = bo[o] + c[b * OC_ + o];
    for (int i = 0; i < OC_; ++i) tv = fmaf(vsh[b][i], Wo[i * OC_ + o], tv);
    tsh[b][o] = tv;
    __syncthreads();

    float m = 0.f;
    for (int i = 0; i < OC_; ++i) m += tsh[b][i];
    m *= (1.f / OC_);
    float var = 0.f;
    for (int i = 0; i < OC_; ++i) { float d = tsh[b][i] - m; var = fmaf(d, d, var); }
    var *= (1.f / OC_);

    const float a = (tv - m) * rsqrtf(var + 1e-5f) * ln_g[o] + ln_b[o];
    const float c2v = c[b * OC_ + o] + a;
    c2ws[b * OC_ + o]  = c2v;
    c2out[b * OC_ + o] = c2v;
}

// ---------------------------------------------------------------------------
// Kernel 2: transpose + bf16-convert the MLP weights so B-fragments are
// contiguous per lane:  W1t[n][k] = bf16(W1[k][n]),  W2t[n][k] = bf16(W2[k][n])
// ---------------------------------------------------------------------------
__global__ __launch_bounds__(256) void prep_weights(
    const float* __restrict__ W1, const float* __restrict__ W2,
    __bf16* __restrict__ W1t, __bf16* __restrict__ W2t)
{
    const int i = blockIdx.x * 256 + threadIdx.x;
    if (i < HID_ * KIN_) {
        const int n = i / KIN_;
        const int k = i - n * KIN_;
        W1t[i] = (__bf16)W1[k * HID_ + n];
    }
    if (i < CH_ * HID_) {
        const int n = i >> 6;
        const int k = i & 63;
        W2t[i] = (__bf16)W2[k * CH_ + n];
    }
}

// ---------------------------------------------------------------------------
// Kernel 3: fused perception (4 dilated depthwise 3x3 + passthrough) and
// update MLP (640->64 GELU -> 64->64) via v_wmma_f32_16x16x32_bf16.
// One wave = 16 pixels of one row; 4 waves / 128-thread block.
// ---------------------------------------------------------------------------
__global__ __launch_bounds__(128) void nca_fused_kernel(
    const float* __restrict__ x,
    const float* __restrict__ pk,
    const float* __restrict__ b1,
    const float* __restrict__ b2,
    const float* __restrict__ c2g,
    const __bf16* __restrict__ W1t,
    const __bf16* __restrict__ W2t,
    float* __restrict__ out)
{
    __shared__ float  pk_s[4 * NCAC_ * 9];               // 18 KB
    __shared__ float  c2_s[B_ * OC_];                    //  2 KB
    __shared__ __attribute__((aligned(16))) __bf16 feat[WAVES_][16][FPAD_]; // ~17 KB
    __shared__ __attribute__((aligned(16))) __bf16 hid [WAVES_][16][HPAD_]; //  ~9 KB

    const int tid = threadIdx.x;
    for (int i = tid; i < 4 * NCAC_ * 9; i += 128) pk_s[i] = pk[i];
    for (int i = tid; i < B_ * OC_;      i += 128) c2_s[i] = c2g[i];
    __syncthreads();

    const int wave = tid >> 5;
    const int lane = tid & 31;
    const int l15  = lane & 15;
    const int hsel = (lane >> 4) & 1;

    const int tile = blockIdx.x * WAVES_ + wave;   // 8192 tiles total
    const int p0   = tile << 4;
    const int b    = p0 >> 14;                     // / (H*W)
    const int rem  = p0 & (H_ * W_ - 1);
    const int py   = rem >> 7;
    const int px0  = rem & (W_ - 1);               // multiple of 16

    const float* __restrict__ xb  = x + (size_t)b * (H_ * W_ * CH_);
    const float* c2b = &c2_s[b * OC_];

    v8f acc[4];
#pragma unroll
    for (int nt = 0; nt < 4; ++nt)
#pragma unroll
        for (int r = 0; r < 8; ++r) acc[nt][r] = 0.f;

#pragma unroll 1
    for (int g = 0; g < 5; ++g) {
        // ---- build one 16x128 bf16 feature slab in LDS ----
        for (int it = 0; it < 64; ++it) {
            const int m  = it >> 2;
            const int ch = ((it & 3) << 5) + lane;
            const int xp = px0 + m;
            float val;
            if (g == 0) {
                val = (ch < CH_) ? xb[(size_t)(py * W_ + xp) * CH_ + ch]
                                 : c2b[ch - CH_];
            } else {
                const int d = 1 << (g - 1);                 // DILS = 1,2,4,8
                const float* __restrict__ kw = &pk_s[((g - 1) * NCAC_ + ch) * 9];
                if (ch < CH_) {
                    float a = 0.f;
#pragma unroll
                    for (int ky = 0; ky < 3; ++ky) {
                        const int yy = py + (ky - 1) * d;
                        if (yy >= 0 && yy < H_) {
                            const float* __restrict__ row =
                                xb + (size_t)(yy * W_) * CH_ + ch;
#pragma unroll
                            for (int kx = 0; kx < 3; ++kx) {
                                const int xx = xp + (kx - 1) * d;
                                if (xx >= 0 && xx < W_)
                                    a = fmaf(kw[ky * 3 + kx],
                                             row[(size_t)xx * CH_], a);
                            }
                        }
                    }
                    val = a;
                } else {
                    // depthwise conv of the spatially-constant context half:
                    // c2 * (sum of in-bounds kernel taps)
                    float s = 0.f;
#pragma unroll
                    for (int ky = 0; ky < 3; ++ky) {
                        const int yy = py + (ky - 1) * d;
                        if (yy >= 0 && yy < H_) {
#pragma unroll
                            for (int kx = 0; kx < 3; ++kx) {
                                const int xx = xp + (kx - 1) * d;
                                if (xx >= 0 && xx < W_) s += kw[ky * 3 + kx];
                            }
                        }
                    }
                    val = c2b[ch - CH_] * s;
                }
            }
            feat[wave][m][ch] = (__bf16)val;
        }
        __syncthreads();

        // ---- 4 WMMA K-steps of 32 against the 4 N-tiles of W1 ----
        const __bf16* frow = &feat[wave][l15][0];
#pragma unroll
        for (int ks = 0; ks < 4; ++ks) {
            const int k0 = ks << 5;
            v8bf alo = *(const v8bf*)(frow + k0 + hsel * 8);
            v8bf ahi = *(const v8bf*)(frow + k0 + 16 + hsel * 8);
            v16bf afrag;
#pragma unroll
            for (int i = 0; i < 8; ++i) { afrag[i] = alo[i]; afrag[8 + i] = ahi[i]; }
            const int kg = (g << 7) + k0;
#pragma unroll
            for (int nt = 0; nt < 4; ++nt) {
                const __bf16* wr =
                    W1t + (size_t)((nt << 4) + l15) * KIN_ + kg + hsel * 8;
                v8bf blo = *(const v8bf*)(wr);
                v8bf bhi = *(const v8bf*)(wr + 16);
                v16bf bfrag;
#pragma unroll
                for (int i = 0; i < 8; ++i) { bfrag[i] = blo[i]; bfrag[8 + i] = bhi[i]; }
                acc[nt] = __builtin_amdgcn_wmma_f32_16x16x32_bf16(
                    false, afrag, false, bfrag, (short)0, acc[nt], false, false);
            }
        }
        __syncthreads();
    }

    // ---- bias + exact GELU, relayout hidden[16][64] through LDS ----
#pragma unroll
    for (int nt = 0; nt < 4; ++nt) {
        const float bias = b1[(nt << 4) + l15];
#pragma unroll
        for (int r = 0; r < 8; ++r) {
            float h = acc[nt][r] + bias;
            h = 0.5f * h * (1.f + erff(h * 0.70710678118654752f));
            hid[wave][r + (hsel << 3)][(nt << 4) + l15] = (__bf16)h;
        }
    }
    __syncthreads();

    // ---- layer 2: hidden[16x64] @ W2[64x64] ----
    v8f oacc[4];
#pragma unroll
    for (int nt = 0; nt < 4; ++nt)
#pragma unroll
        for (int r = 0; r < 8; ++r) oacc[nt][r] = 0.f;

    const __bf16* hrow = &hid[wave][l15][0];
#pragma unroll
    for (int ks = 0; ks < 2; ++ks) {
        const int k0 = ks << 5;
        v8bf alo = *(const v8bf*)(hrow + k0 + hsel * 8);
        v8bf ahi = *(const v8bf*)(hrow + k0 + 16 + hsel * 8);
        v16bf afrag;
#pragma unroll
        for (int i = 0; i < 8; ++i) { afrag[i] = alo[i]; afrag[8 + i] = ahi[i]; }
#pragma unroll
        for (int nt = 0; nt < 4; ++nt) {
            const __bf16* wr =
                W2t + (size_t)((nt << 4) + l15) * HID_ + k0 + hsel * 8;
            v8bf blo = *(const v8bf*)(wr);
            v8bf bhi = *(const v8bf*)(wr + 16);
            v16bf bfrag;
#pragma unroll
            for (int i = 0; i < 8; ++i) { bfrag[i] = blo[i]; bfrag[8 + i] = bhi[i]; }
            oacc[nt] = __builtin_amdgcn_wmma_f32_16x16x32_bf16(
                false, afrag, false, bfrag, (short)0, oacc[nt], false, false);
        }
    }

    // ---- bias + store y (f32, [B,H,W,64]) ----
    float* __restrict__ ob =
        out + (size_t)(b * (H_ * W_) + py * W_ + px0) * CH_;
#pragma unroll
    for (int nt = 0; nt < 4; ++nt) {
        const float bias = b2[(nt << 4) + l15];
#pragma unroll
        for (int r = 0; r < 8; ++r) {
            const int m = r + (hsel << 3);
            ob[(size_t)m * CH_ + (nt << 4) + l15] = oacc[nt][r] + bias;
        }
    }
}

// ---------------------------------------------------------------------------
extern "C" void kernel_launch(void* const* d_in, const int* in_sizes, int n_in,
                              void* d_out, int out_size, void* d_ws, size_t ws_size,
                              hipStream_t stream) {
    (void)in_sizes; (void)n_in; (void)out_size; (void)ws_size;
    const float* x    = (const float*)d_in[0];
    const float* c    = (const float*)d_in[1];
    // d_in[2..5] = Wq,bq,Wk,bk: provably unused (attended == v)
    const float* Wv   = (const float*)d_in[6];
    const float* bv   = (const float*)d_in[7];
    const float* Wo   = (const float*)d_in[8];
    const float* bo   = (const float*)d_in[9];
    const float* ln_g = (const float*)d_in[10];
    const float* ln_b = (const float*)d_in[11];
    const float* pk   = (const float*)d_in[12];
    const float* W1   = (const float*)d_in[13];
    const float* b1   = (const float*)d_in[14];
    const float* W2   = (const float*)d_in[15];
    const float* b2   = (const float*)d_in[16];

    float* y   = (float*)d_out;
    float* c2o = y + (size_t)B_ * H_ * W_ * CH_;   // second tuple output

    float*  c2w = (float*)d_ws;
    __bf16* W1t = (__bf16*)((char*)d_ws + 4096);
    __bf16* W2t = (__bf16*)((char*)d_ws + 4096 + 2 * HID_ * KIN_);

    hipLaunchKernelGGL(ctx_kernel, dim3(1), dim3(512), 0, stream,
                       c, Wv, bv, Wo, bo, ln_g, ln_b, c2w, c2o);
    hipLaunchKernelGGL(prep_weights, dim3(160), dim3(256), 0, stream,
                       W1, W2, W1t, W2t);
    const int tiles  = (B_ * H_ * W_) / 16;        // 8192
    const int blocks = tiles / WAVES_;             // 2048
    hipLaunchKernelGGL(nca_fused_kernel, dim3(blocks), dim3(128), 0, stream,
                       x, pk, b1, b2, c2w, W1t, W2t, y);
}
